// DotProductAttention_62886911148691
// MI455X (gfx1250) — compile-verified, matrix operand order checked by reading, not run on hardware
//
#include <hip/hip_runtime.h>

// Problem constants (from reference): B=4, N=16, L=1024, D=64
#define BB 4
#define NN 16
#define LL 1024
#define DD 64
#define TI 64            // query rows per workgroup
#define SSTR 1028        // score-tile LDS row stride (floats), padded vs 1024
#define QSTR 72          // Q tile LDS row stride (bf16), padded vs 64
#define PSTR 35          // per-wave pos-scratch row stride (floats)
#define VSTR 72          // V tile LDS row stride (bf16)

typedef __attribute__((ext_vector_type(16))) __bf16 v16bf;
typedef __attribute__((ext_vector_type(8)))  float  v8f;

// Dynamic-LDS carve (bytes). Total 304128 <= 320KB WGP LDS.
#define OFF_S   0
#define SZ_S    (TI * SSTR * 4)        // 263168: 64x1024 f32 score tile (+pad)
#define OFF_QC  (OFF_S + SZ_S)
#define SZ_Q    (TI * QSTR * 2)        // 9216: 64x64 bf16
#define OFF_QP  (OFF_QC + SZ_Q)
#define OFF_PS  (OFF_QP + SZ_Q)
#define SZ_PS   (8 * 16 * PSTR * 4)    // 17920: per-wave 16x35 f32 scratch
#define OFF_VT  (OFF_PS + SZ_PS)
#define SZ_VT   (32 * VSTR * 2)        // 4608: 32x64 bf16 V chunk
#define SMEM_BYTES (OFF_VT + SZ_VT)

// --- WMMA helper: D = A(16x32 bf16) x B(32x16 bf16) + C(16x16 f32) ---
__device__ __forceinline__ v8f wmma_bf16(v16bf a, v16bf b, v8f c) {
  return __builtin_amdgcn_wmma_f32_16x16x32_bf16(false, a, false, b,
                                                 (short)0, c, false, false);
}

// A-fragment (16x32 bf16) from a row-major bf16 LDS tile.
// lane<16 holds K = off+{0..7} and off+{16..23}; lane>=16 uses off = base+8.
__device__ __forceinline__ v16bf afrag_bf16(const __bf16* row, int off) {
  v16bf a;
#pragma unroll
  for (int e = 0; e < 8; ++e) { a[e] = row[off + e]; a[e + 8] = row[off + 16 + e]; }
  return a;
}

// A-fragment from a row-major f32 LDS tile (convert to bf16 on the fly).
__device__ __forceinline__ v16bf afrag_f32(const float* row, int off) {
  v16bf a;
#pragma unroll
  for (int e = 0; e < 8; ++e) {
    a[e]     = (__bf16)row[off + e];
    a[e + 8] = (__bf16)row[off + 16 + e];
  }
  return a;
}

// B-fragment (32x16, B = Xrow^T) built from one global f32 row: lane owns
// column n = row index of X; identical K striping as the A fragment.
__device__ __forceinline__ v16bf bfrag_gmem(const float* __restrict__ row, int off) {
  v16bf b;
#pragma unroll
  for (int e = 0; e < 8; ++e) {
    b[e]     = (__bf16)row[off + e];
    b[e + 8] = (__bf16)row[off + 16 + e];
  }
  return b;
}

// B-fragment for V (no transpose): B[k][n] = V[kbase+k][col], column gather
// from the staged bf16 V chunk in LDS.
__device__ __forceinline__ v16bf bfrag_v(const __bf16* Vt, int half, int col) {
  v16bf b;
#pragma unroll
  for (int e = 0; e < 8; ++e) {
    b[e]     = Vt[(half + e) * VSTR + col];
    b[e + 8] = Vt[(half + 16 + e) * VSTR + col];
  }
  return b;
}

__global__ void __launch_bounds__(256)
relattn_kernel(const float* __restrict__ qc, const float* __restrict__ qp,
               const float* __restrict__ key, const float* __restrict__ val,
               const float* __restrict__ pos,
               float* __restrict__ out, float* __restrict__ wgt) {
  extern __shared__ char smem[];
  float*  S  = (float*)(smem + OFF_S);
  __bf16* Qc = (__bf16*)(smem + OFF_QC);
  __bf16* Qp = (__bf16*)(smem + OFF_QP);
  float*  PS = (float*)(smem + OFF_PS);
  __bf16* Vt = (__bf16*)(smem + OFF_VT);

  const int tid  = threadIdx.x;
  const int wave = tid >> 5;
  const int lane = tid & 31;
  const int ln16 = lane & 15;
  const int half = (lane < 16) ? 0 : 8;   // K-striping offset per lane half

  const int tileId = blockIdx.x & 15;     // L/TI = 16 row tiles
  const int bn     = blockIdx.x >> 4;     // 0..B*N-1
  const int n      = bn & (NN - 1);
  const int i0     = tileId * TI;

  const float* qc_g = qc  + (size_t)bn * LL * DD + (size_t)i0 * DD;
  const float* qp_g = qp  + (size_t)bn * LL * DD + (size_t)i0 * DD;
  const float* k_g  = key + (size_t)bn * LL * DD;
  const float* v_g  = val + (size_t)bn * LL * DD;
  const float* p_g  = pos + (size_t)n  * LL * DD;
  float* o_g = out + (size_t)bn * LL * DD;
  float* w_g = wgt + (size_t)bn * LL * LL;

  // ---------------- Phase 0: stage Q tiles as bf16 in LDS ----------------
  for (int idx = tid; idx < TI * DD; idx += 256) {
    int r = idx >> 6, c = idx & 63;
    Qc[r * QSTR + c] = (__bf16)qc_g[idx];
    Qp[r * QSTR + c] = (__bf16)qp_g[idx];
  }
  __syncthreads();

  // ---------------- Phase 1: S = (Qc K^T + rel_shift(Qp P^T)) / 8 --------
  // 16-col tiles 0..(i0+63)/16 round-robin across the 8 waves.
  const int njt = (i0 + TI) >> 4;
  float* sc = PS + wave * 16 * PSTR;      // per-wave scratch (16 x 35 f32)
  for (int jt = wave; jt < njt; jt += 8) {
    const int j0c = jt << 4;
    const float* krow = k_g + (size_t)(j0c + ln16) * DD;  // lane owns key row
    v16bf kf0 = bfrag_gmem(krow, half);
    v16bf kf1 = bfrag_gmem(krow, half + 32);
#pragma unroll
    for (int ir = 0; ir < 4; ++ir) {
      const int i0r = i0 + ir * 16;
      const __bf16* qcrow = Qc + (ir * 16 + ln16) * QSTR;
      const __bf16* qprow = Qp + (ir * 16 + ln16) * QSTR;
      // content scores
      v16bf a0 = afrag_bf16(qcrow, half);
      v16bf a1 = afrag_bf16(qcrow, half + 32);
      v8f c = {};
      c = wmma_bf16(a0, kf0, c);
      c = wmma_bf16(a1, kf1, c);
      // positional band: M[ii][t] = Qp[i0r+ii] . P[pstart+t], t in [0,31]
      // rel_shift gather: S_pos[ii][jj] = M[ii][15 - ii + jj]
      const int pstart = LL - 16 - i0r + j0c;
      int pr0 = pstart + ln16;        if (pr0 < 0) pr0 = 0; if (pr0 > LL - 1) pr0 = LL - 1;
      int pr1 = pstart + 16 + ln16;   if (pr1 < 0) pr1 = 0; if (pr1 > LL - 1) pr1 = LL - 1;
      const float* p0 = p_g + (size_t)pr0 * DD;
      const float* p1 = p_g + (size_t)pr1 * DD;
      v16bf ap0 = afrag_bf16(qprow, half);
      v16bf ap1 = afrag_bf16(qprow, half + 32);
      v16bf pb00 = bfrag_gmem(p0, half);
      v16bf pb01 = bfrag_gmem(p0, half + 32);
      v16bf pb10 = bfrag_gmem(p1, half);
      v16bf pb11 = bfrag_gmem(p1, half + 32);
      v8f m0 = {}, m1 = {};
      m0 = wmma_bf16(ap0, pb00, m0);
      m0 = wmma_bf16(ap1, pb01, m0);
      m1 = wmma_bf16(ap0, pb10, m1);
      m1 = wmma_bf16(ap1, pb11, m1);
      // bounce M through per-wave LDS scratch for the diagonal gather
#pragma unroll
      for (int r = 0; r < 8; ++r) {
        int mm = r + half;                        // C/D row = reg + half
        sc[mm * PSTR + ln16]      = m0[r];
        sc[mm * PSTR + 16 + ln16] = m1[r];
      }
      asm volatile("s_wait_dscnt 0" ::: "memory");  // wave-internal DS order
#pragma unroll
      for (int r = 0; r < 8; ++r) {
        int mm = r + half;
        float pv = sc[mm * PSTR + (15 - mm + ln16)];
        S[(ir * 16 + mm) * SSTR + j0c + ln16] = (c[r] + pv) * 0.125f;
      }
    }
  }
  __syncthreads();

  // ---------------- Phase 2: causal softmax, emit W ----------------------
  {
    const int row = tid >> 2, sub = tid & 3;     // 4 lanes per query row
    const int i = i0 + row;
    float* srow = S + row * SSTR;
    float mx = -3.0e38f;
    for (int j = sub; j <= i; j += 4) mx = fmaxf(mx, srow[j]);
    mx = fmaxf(mx, __shfl_xor(mx, 1, 32));
    mx = fmaxf(mx, __shfl_xor(mx, 2, 32));
    float sum = 0.f;
    for (int j = sub; j <= i; j += 4) {
      float e = __expf(srow[j] - mx);
      srow[j] = e;
      sum += e;
    }
    sum += __shfl_xor(sum, 1, 32);
    sum += __shfl_xor(sum, 2, 32);
    const float inv = 1.0f / sum;
    float* wrow = w_g + (size_t)i * LL;
    for (int j = sub; j < LL; j += 4) {
      if (j <= i) {
        float w = srow[j] * inv;
        srow[j] = w;          // keep normalized W in LDS for phase 3
        wrow[j] = w;
      } else {
        wrow[j] = 0.f;        // masked weights are exactly zero
        if (j < i0 + TI) srow[j] = 0.f;   // zero LDS columns phase 3 reads
      }
    }
  }
  __syncthreads();

  // ---------------- Phase 3: O = W @ V ----------------------------------
  // 16 output tiles (4 ir x 4 ic) over 8 waves -> 2 row-tiles per wave.
  const int ic  = wave & 3;
  const int irb = (wave >> 2) * 2;
  const int col = ic * 16 + ln16;
  v8f o0 = {}, o1 = {};
  const int nkb = (i0 + TI) >> 5;         // W columns beyond i0+63 are zero
  for (int kb = 0; kb < nkb; ++kb) {
    __syncthreads();
    for (int idx = tid; idx < 32 * DD; idx += 256) {
      int r = idx >> 6, cc = idx & 63;
      Vt[r * VSTR + cc] = (__bf16)v_g[(size_t)(kb * 32 + r) * DD + cc];
    }
    __syncthreads();
    v16bf a0 = afrag_f32(S + (irb * 16 + ln16) * SSTR + kb * 32, half);
    v16bf a1 = afrag_f32(S + ((irb + 1) * 16 + ln16) * SSTR + kb * 32, half);
    v16bf bv = bfrag_v(Vt, half, col);
    o0 = wmma_bf16(a0, bv, o0);
    o1 = wmma_bf16(a1, bv, o1);
  }
#pragma unroll
  for (int r = 0; r < 8; ++r) {
    int mm = r + half;
    o_g[(size_t)(i0 + irb * 16 + mm) * DD + col]       = o0[r];
    o_g[(size_t)(i0 + (irb + 1) * 16 + mm) * DD + col] = o1[r];
  }
}

extern "C" void kernel_launch(void* const* d_in, const int* in_sizes, int n_in,
                              void* d_out, int out_size, void* d_ws, size_t ws_size,
                              hipStream_t stream) {
  (void)in_sizes; (void)n_in; (void)d_ws; (void)ws_size; (void)out_size;
  const float* qc  = (const float*)d_in[0];
  const float* qp  = (const float*)d_in[1];
  const float* key = (const float*)d_in[2];
  const float* val = (const float*)d_in[3];
  const float* pos = (const float*)d_in[4];
  // d_in[5] is the causal mask; it is triu(k=1), known at compile time.
  float* out = (float*)d_out;                       // [B,N,L,D]
  float* wgt = out + (size_t)BB * NN * LL * DD;     // [B,N,L,L]
  dim3 grid(BB * NN * (LL / TI));                   // 1024 workgroups
  relattn_kernel<<<grid, 256, SMEM_BYTES, stream>>>(qc, qp, key, val, pos, out, wgt);
}
